// MoEAutoregressiveVM_84000970375603
// MI455X (gfx1250) — matrix operand order repaired and checked by reading, not compiled, not run on hardware
//
#include <hip/hip_runtime.h>
#include <stdint.h>

typedef unsigned short u16;
typedef __attribute__((ext_vector_type(16))) __bf16 v16bf;
typedef __attribute__((ext_vector_type(8)))  float  v8f;

namespace cfg {
constexpr int S    = 1024;
constexpr int D    = 1352;
constexpr int Dp   = 1408;   // D padded to multiple of 64
constexpr int H    = 8;
constexpr int HD   = 169;
constexpr int HDp  = 192;    // head dim padded (mult of 64)
constexpr int TD   = 3 * D;  // 4056
constexpr int TDp  = 4096;   // 3D padded
constexpr int DFF  = 2704;
constexpr int DFFp = 2752;   // DFF padded
constexpr int E    = 4;
constexpr int L    = 7;
constexpr int NUM_OPS = 16;
}

__device__ __forceinline__ u16 f32_to_bf16(float f) {
  unsigned u = __float_as_uint(f);
  u += 0x7FFFu + ((u >> 16) & 1u);   // round to nearest even
  return (u16)(u >> 16);
}

union FragBF { v16bf v; uint4 u[2]; };

// CDNA5 async global->LDS copy of 16 bytes (tracked by ASYNCcnt).
// VDST = wave-relative LDS byte address (low 32 bits of generic pointer),
// VADDR = 64-bit global address (GV mode).
__device__ __forceinline__ void async_cp16(void* lds_dst, const void* gsrc) {
  unsigned loff = (unsigned)(uintptr_t)lds_dst;
  unsigned long long ga = (unsigned long long)(uintptr_t)gsrc;
  asm volatile("global_load_async_to_lds_b128 %0, %1, off"
               :: "v"(loff), "v"(ga) : "memory");
}
__device__ __forceinline__ void wait_async0() {
  asm volatile("s_wait_asynccnt 0x0" ::: "memory");
}

// ---------------------------------------------------------------------------
// bf16 WMMA GEMM (NT):  C[z] = A[z] * B[z]^T (+bias) (+C)   (f32 accumulate)
//   A: row-major M x K (bf16).  B: row-major N x K (bf16)  -- pre-transposed.
//   Tile 64x64x32; 128 threads = 4 waves, each wave 32x32 (2x2 wmma tiles).
//   Double-buffered LDS staging via async global->LDS loads (ASYNCcnt).
//   Dynamic row count via Mdyn (device pointer) for MoE expert segments.
// ---------------------------------------------------------------------------
template <bool RELU, bool ACC, bool OUT_BF16>
__global__ __launch_bounds__(128) void gemm_nt_wmma(
    const u16* __restrict__ A, int lda, long sA,
    const u16* __restrict__ B, int ldb, long sB,
    void* __restrict__ Cv, int ldc, long sC,
    const float* __restrict__ bias,
    int K, int Mstatic, const int* __restrict__ Mdyn, int Nstore)
{
  constexpr int LDT = 40;  // LDS row stride (u16); 80B = 5*16B, keeps 16B align
  __shared__ __align__(16) u16 As[2][64 * LDT];
  __shared__ __align__(16) u16 Bs[2][64 * LDT];

  const int z = blockIdx.z;
  const int Mlim = Mdyn ? Mdyn[0] : Mstatic;
  const int m0 = blockIdx.y * 64;
  if (m0 >= Mlim) return;              // whole block exits uniformly
  const int n0 = blockIdx.x * 64;
  const u16* Ab = A + (long)z * sA;
  const u16* Bb = B + (long)z * sB;

  const int t    = threadIdx.x;
  const int lane = t & 31;
  const int wave = t >> 5;
  const int wy   = wave >> 1;
  const int wx   = wave & 1;

  v8f acc[2][2];
  #pragma unroll
  for (int i = 0; i < 2; ++i)
    #pragma unroll
    for (int j = 0; j < 2; ++j)
      acc[i][j] = (v8f)(0.0f);

  // ISA fragment lane mapping (16-bit A 16x32 / B 32x16, wave32)
  const int rsel = lane & 15;
  const int kh   = (lane < 16) ? 0 : 8;    // A: K half offset
  const int ks   = (lane < 16) ? 0 : 8;    // B: K half offset

  // stage one 64x32 A tile + 64x32 B tile into buffer `buf` (async)
  auto stage = [&](int k0, int buf) {
    #pragma unroll
    for (int i = 0; i < 2; ++i) {
      int c  = i * 128 + t;      // 256 chunks of 8 bf16 each
      int r  = c >> 2;
      int co = (c & 3) * 8;
      async_cp16(&As[buf][r * LDT + co], Ab + (long)(m0 + r) * lda + k0 + co);
      async_cp16(&Bs[buf][r * LDT + co], Bb + (long)(n0 + r) * ldb + k0 + co);
    }
  };

  stage(0, 0);
  for (int k0 = 0; k0 < K; k0 += 32) {
    const int cur = (k0 >> 5) & 1;
    wait_async0();        // our async loads into buf `cur` are in LDS
    __syncthreads();      // everyone's loads done; everyone's prior reads done
    if (k0 + 32 < K) stage(k0 + 32, cur ^ 1);   // prefetch next slice

    FragBF af[2], bfr[2];
    #pragma unroll
    for (int i = 0; i < 2; ++i) {
      int r = wy * 32 + i * 16 + rsel;
      af[i].u[0] = *(const uint4*)(&As[cur][r * LDT + kh]);        // K kh..kh+7
      af[i].u[1] = *(const uint4*)(&As[cur][r * LDT + kh + 16]);   // K kh+16..+23
    }
    #pragma unroll
    for (int j = 0; j < 2; ++j) {
      int n = wx * 32 + j * 16 + rsel;
      bfr[j].u[0] = *(const uint4*)(&Bs[cur][n * LDT + ks]);       // K ks..ks+7
      bfr[j].u[1] = *(const uint4*)(&Bs[cur][n * LDT + ks + 16]);  // K ks+16..+23
    }
    #pragma unroll
    for (int i = 0; i < 2; ++i)
      #pragma unroll
      for (int j = 0; j < 2; ++j)
        acc[i][j] = __builtin_amdgcn_wmma_f32_16x16x32_bf16(
            false, af[i].v, false, bfr[j].v, (short)0, acc[i][j], false, false);
  }

  // ---- store (C layout: lanes 0-15 -> rows 0-7, lanes 16-31 -> rows 8-15) ----
  const int nlane = lane & 15;
  const int rb    = (lane < 16) ? 0 : 8;
  #pragma unroll
  for (int i = 0; i < 2; ++i) {
    #pragma unroll
    for (int j = 0; j < 2; ++j) {
      int col = n0 + wx * 32 + j * 16 + nlane;
      if (col >= Nstore) continue;
      float bv = bias ? bias[col] : 0.0f;
      #pragma unroll
      for (int r = 0; r < 8; ++r) {
        int row = m0 + wy * 32 + i * 16 + rb + r;
        if (row >= Mlim) continue;
        float val = acc[i][j][r] + bv;
        if (RELU) val = fmaxf(val, 0.0f);
        long off = (long)z * sC + (long)row * ldc + col;
        if (OUT_BF16) {
          ((u16*)Cv)[off] = f32_to_bf16(val);
        } else {
          float* C = (float*)Cv;
          if (ACC) C[off] += val;
          else     C[off] = val;
        }
      }
    }
  }
}

// ---------------------------------------------------------------------------
// Routing: op = argmax(x[s,0:16]) (first max, like jnp.argmax), expert = op%4.
// Token s gets slot = e*S + local_index within its padded per-expert segment.
// ---------------------------------------------------------------------------
__global__ __launch_bounds__(cfg::S) void route_kernel(
    const float* __restrict__ x, int* __restrict__ slot, int* __restrict__ cnt)
{
  using namespace cfg;
  __shared__ int scnt[E];
  int t = threadIdx.x;
  if (t < E) scnt[t] = 0;
  __syncthreads();
  const float* row = x + (long)t * D;
  float best = row[0]; int bi = 0;
  #pragma unroll
  for (int j = 1; j < NUM_OPS; ++j) {
    float v = row[j];
    if (v > best) { best = v; bi = j; }
  }
  int e  = bi & (E - 1);
  int li = atomicAdd(&scnt[e], 1);
  __syncthreads();
  slot[t] = e * S + li;
  if (t < E) cnt[t] = scnt[t];
}

// LayerNorm(f32) -> bf16 output, zero-padded to Dpad columns
__global__ __launch_bounds__(256) void layernorm_to_bf16(
    const float* __restrict__ x, const float* __restrict__ g,
    const float* __restrict__ b, u16* __restrict__ out, int Dn, int Dpad)
{
  int row = blockIdx.x, tid = threadIdx.x;
  const float* src = x + (long)row * Dn;
  u16* dst = out + (long)row * Dpad;
  __shared__ float red[256];
  float s = 0.f;
  for (int d = tid; d < Dn; d += 256) s += src[d];
  red[tid] = s; __syncthreads();
  for (int st = 128; st > 0; st >>= 1) { if (tid < st) red[tid] += red[tid + st]; __syncthreads(); }
  float mu = red[0] / (float)Dn;
  __syncthreads();
  float v = 0.f;
  for (int d = tid; d < Dn; d += 256) { float t2 = src[d] - mu; v += t2 * t2; }
  red[tid] = v; __syncthreads();
  for (int st = 128; st > 0; st >>= 1) { if (tid < st) red[tid] += red[tid + st]; __syncthreads(); }
  float rstd = rsqrtf(red[0] / (float)Dn + 1e-5f);
  for (int d = tid; d < Dpad; d += 256)
    dst[d] = (d < Dn) ? f32_to_bf16((src[d] - mu) * rstd * g[d] + b[d]) : (u16)0;
}

// f32 [Kr,Nr] (batched) -> TRANSPOSED bf16 [Np,Kp] zero-padded (B^T for NT GEMM)
__global__ __launch_bounds__(256) void convert_pad_t(
    const float* __restrict__ in, u16* __restrict__ out,
    int Kr, int Nr, int Kp, int Np, long sIn, long sOut)
{
  int idx = blockIdx.x * 256 + threadIdx.x;
  if (idx >= Np * Kp) return;
  long zb = (long)blockIdx.z;
  int k = idx % Kp, n = idx / Kp;
  out[zb * sOut + idx] = (k < Kr && n < Nr) ? f32_to_bf16(in[zb * sIn + (long)k * Nr + n]) : (u16)0;
}

__global__ __launch_bounds__(256) void pad_bias_kernel(
    const float* __restrict__ in, float* __restrict__ out, int Nr, int Np, int count)
{
  int idx = blockIdx.x * 256 + threadIdx.x;
  if (idx >= count * Np) return;
  int n = idx % Np, e = idx / Np;
  out[idx] = (n < Nr) ? in[(long)e * Nr + n] : 0.f;
}

__global__ __launch_bounds__(256) void zero_u32(unsigned* __restrict__ p, int n)
{
  int i = blockIdx.x * 256 + threadIdx.x;
  if (i < n) p[i] = 0u;
}

// qkv f32 [S, TDp(ld)] -> Q,K bf16 [H][S][HDp] (zero-padded head dim),
//                         V  bf16 [H][HDp][S]  (head-transposed, for NT GEMM)
__global__ __launch_bounds__(256) void qkv_split_kernel(
    const float* __restrict__ qkv, u16* __restrict__ Q,
    u16* __restrict__ Kk, u16* __restrict__ Vt)
{
  using namespace cfg;
  int idx = blockIdx.x * 256 + threadIdx.x;
  if (idx >= S * H * HDp) return;
  int d  = idx % HDp;
  int hh = (idx / HDp) % H;
  int s  = idx / (HDp * H);
  long o  = ((long)hh * S + s) * HDp + d;
  long ov = ((long)hh * HDp + d) * S + s;
  if (d < HD) {
    int col = hh * HD + d;
    const float* row = qkv + (long)s * TDp;
    Q[o]   = f32_to_bf16(row[col]);
    Kk[o]  = f32_to_bf16(row[D + col]);
    Vt[ov] = f32_to_bf16(row[2 * D + col]);
  } else {
    Q[o] = 0; Kk[o] = 0; Vt[ov] = 0;
  }
}

// causal softmax over scores[h*S+i][:] (scaled), bf16 out, zeros for j>i
__global__ __launch_bounds__(256) void softmax_causal(
    const float* __restrict__ scores, u16* __restrict__ attn, float scale)
{
  using namespace cfg;
  int r = blockIdx.x;            // h*S + i
  int i = r % S;
  const float* src = scores + (long)r * S;
  u16* dst = attn + (long)r * S;
  __shared__ float red[256];
  int tid = threadIdx.x;
  float mx = -1e30f;
  for (int j = tid; j <= i; j += 256) mx = fmaxf(mx, src[j] * scale);
  red[tid] = mx; __syncthreads();
  for (int st = 128; st > 0; st >>= 1) { if (tid < st) red[tid] = fmaxf(red[tid], red[tid + st]); __syncthreads(); }
  mx = red[0]; __syncthreads();
  float sum = 0.f;
  for (int j = tid; j <= i; j += 256) sum += __expf(src[j] * scale - mx);
  red[tid] = sum; __syncthreads();
  for (int st = 128; st > 0; st >>= 1) { if (tid < st) red[tid] += red[tid + st]; __syncthreads(); }
  float inv = 1.f / red[0];
  for (int j = tid; j < S; j += 256)
    dst[j] = (j <= i) ? f32_to_bf16(__expf(src[j] * scale - mx) * inv) : (u16)0;
}

// gather bf16 rows (as u32 pairs) into per-expert segments via slot[]
__global__ __launch_bounds__(256) void gather_rows(
    const unsigned* __restrict__ src, unsigned* __restrict__ dst, const int* __restrict__ slot)
{
  using namespace cfg;
  constexpr int RW = Dp / 2;
  int idx = blockIdx.x * 256 + threadIdx.x;
  if (idx >= S * RW) return;
  int t = idx / RW, d2 = idx % RW;
  dst[(long)slot[t] * RW + d2] = src[(long)t * RW + d2];
}

// h[t][d] += y[slot[t]][d]
__global__ __launch_bounds__(256) void scatter_add_rows(
    float* __restrict__ h, const float* __restrict__ y, const int* __restrict__ slot)
{
  using namespace cfg;
  int idx = blockIdx.x * 256 + threadIdx.x;
  if (idx >= S * D) return;
  int t = idx / D, d = idx % D;
  h[idx] += y[(long)slot[t] * Dp + d];
}

// ---------------------------------------------------------------------------
extern "C" void kernel_launch(void* const* d_in, const int* in_sizes, int n_in,
                              void* d_out, int out_size, void* d_ws, size_t ws_size,
                              hipStream_t stream)
{
  using namespace cfg;
  const float* x     = (const float*)d_in[0];
  const float* ln1_g = (const float*)d_in[1];
  const float* ln1_b = (const float*)d_in[2];
  const float* wqkv  = (const float*)d_in[3];
  const float* bqkv  = (const float*)d_in[4];
  const float* wo    = (const float*)d_in[5];
  const float* bo    = (const float*)d_in[6];
  const float* ln2_g = (const float*)d_in[7];
  const float* ln2_b = (const float*)d_in[8];
  const float* w1    = (const float*)d_in[9];
  const float* b1    = (const float*)d_in[10];
  const float* w2    = (const float*)d_in[11];
  const float* b2    = (const float*)d_in[12];

  char* wsp = (char*)d_ws;
  auto carve = [&](size_t bytes) -> void* {
    void* p = (void*)wsp;
    wsp += (bytes + 255) & ~(size_t)255;
    return p;
  };

  int*   slot    = (int*)carve((size_t)S * sizeof(int));
  int*   cnt     = (int*)carve((size_t)E * sizeof(int));
  u16*   wqkv_t  = (u16*)carve((size_t)TDp * Dp * 2);   // [TDp][Dp]  (B^T)
  u16*   wo_t    = (u16*)carve((size_t)Dp * Dp * 2);    // [Dp][Dp]
  u16*   w1_t    = (u16*)carve((size_t)E * DFFp * Dp * 2);   // [E][DFFp][Dp]
  u16*   w2_t    = (u16*)carve((size_t)E * Dp * DFFp * 2);   // [E][Dp][DFFp]
  float* b1_pad  = (float*)carve((size_t)E * DFFp * 4);
  u16*   a_bf    = (u16*)carve((size_t)S * Dp * 2);
  float* qkv32   = (float*)carve((size_t)S * TDp * 4);
  u16*   Qh      = (u16*)carve((size_t)H * S * HDp * 2);
  u16*   Kh      = (u16*)carve((size_t)H * S * HDp * 2);
  u16*   Vt      = (u16*)carve((size_t)H * HDp * S * 2);
  float* scores  = (float*)carve((size_t)H * S * S * 4);
  u16*   attn    = (u16*)carve((size_t)H * S * S * 2);
  u16*   obf     = (u16*)carve((size_t)S * Dp * 2);
  u16*   m_bf    = (u16*)carve((size_t)S * Dp * 2);
  u16*   mg_bf   = (u16*)carve((size_t)E * S * Dp * 2);
  u16*   hid_bf  = (u16*)carve((size_t)E * S * DFFp * 2);
  float* y32     = (float*)carve((size_t)E * S * Dp * 4);

  float* h = (float*)d_out;   // residual stream lives in d_out
  hipMemcpyAsync(h, x, (size_t)S * D * sizeof(float), hipMemcpyDeviceToDevice, stream);

  // routing from the ORIGINAL x (fixed across layers, as in the reference)
  route_kernel<<<1, S, 0, stream>>>(x, slot, cnt);
  // zero attention-output buffer once (pad columns must stay zero)
  zero_u32<<<(S * Dp / 2 + 255) / 256, 256, 0, stream>>>((unsigned*)obf, S * Dp / 2);

  const float scale = 1.0f / 13.0f;  // 1/sqrt(169)

  for (int l = 0; l < L; ++l) {
    const float* wqkv_l = wqkv + (size_t)l * D * TD;
    const float* bqkv_l = bqkv + (size_t)l * TD;
    const float* wo_l   = wo   + (size_t)l * D * D;
    const float* bo_l   = bo   + (size_t)l * D;
    const float* w1_l   = w1   + (size_t)l * E * D * DFF;
    const float* b1_l   = b1   + (size_t)l * E * DFF;
    const float* w2_l   = w2   + (size_t)l * E * DFF * D;
    const float* b2_l   = b2   + (size_t)l * E * D;

    // ---- weight conversion to padded, transposed bf16 ----
    convert_pad_t<<<dim3((TDp * Dp + 255) / 256, 1, 1), 256, 0, stream>>>(
        wqkv_l, wqkv_t, D, TD, Dp, TDp, 0, 0);
    convert_pad_t<<<dim3((Dp * Dp + 255) / 256, 1, 1), 256, 0, stream>>>(
        wo_l, wo_t, D, D, Dp, Dp, 0, 0);
    convert_pad_t<<<dim3((DFFp * Dp + 255) / 256, 1, E), 256, 0, stream>>>(
        w1_l, w1_t, D, DFF, Dp, DFFp, (long)D * DFF, (long)DFFp * Dp);
    convert_pad_t<<<dim3((Dp * DFFp + 255) / 256, 1, E), 256, 0, stream>>>(
        w2_l, w2_t, DFF, D, DFFp, Dp, (long)DFF * D, (long)Dp * DFFp);
    pad_bias_kernel<<<(E * DFFp + 255) / 256, 256, 0, stream>>>(b1_l, b1_pad, DFF, DFFp, E);

    // ---- attention block ----
    layernorm_to_bf16<<<S, 256, 0, stream>>>(h, ln1_g + (size_t)l * D, ln1_b + (size_t)l * D, a_bf, D, Dp);

    // qkv = ln1(h) @ wqkv + bqkv
    gemm_nt_wmma<false, false, false><<<dim3(TDp / 64, S / 64, 1), 128, 0, stream>>>(
        a_bf, Dp, 0, wqkv_t, Dp, 0, qkv32, TDp, 0, bqkv_l, Dp, S, nullptr, TD);

    qkv_split_kernel<<<(S * H * HDp + 255) / 256, 256, 0, stream>>>(qkv32, Qh, Kh, Vt);

    // scores[h] = Q_h @ K_h^T   (batched over heads)
    gemm_nt_wmma<false, false, false><<<dim3(S / 64, S / 64, H), 128, 0, stream>>>(
        Qh, HDp, (long)S * HDp, Kh, HDp, (long)S * HDp,
        scores, S, (long)S * S, nullptr, HDp, S, nullptr, S);

    softmax_causal<<<H * S, 256, 0, stream>>>(scores, attn, scale);

    // O_h = attn_h @ V_h  (B = Vt[h] is [HDp][S]) -> bf16 [S,Dp] at col h*HD
    gemm_nt_wmma<false, false, true><<<dim3(HDp / 64, S / 64, H), 128, 0, stream>>>(
        attn, S, (long)S * S, Vt, S, (long)HDp * S,
        obf, Dp, (long)HD, nullptr, S, S, nullptr, HD);

    // h += O @ wo + bo   (accumulate = residual)
    gemm_nt_wmma<false, true, false><<<dim3(Dp / 64, S / 64, 1), 128, 0, stream>>>(
        obf, Dp, 0, wo_t, Dp, 0, h, D, 0, bo_l, Dp, S, nullptr, D);

    // ---- MoE block (routed expert only) ----
    layernorm_to_bf16<<<S, 256, 0, stream>>>(h, ln2_g + (size_t)l * D, ln2_b + (size_t)l * D, m_bf, D, Dp);

    gather_rows<<<(S * (Dp / 2) + 255) / 256, 256, 0, stream>>>(
        (const unsigned*)m_bf, (unsigned*)mg_bf, slot);

    for (int e = 0; e < E; ++e) {
      // hid = relu(mg_e @ W1_e + b1_e)   (bf16 out, dynamic M = cnt[e])
      gemm_nt_wmma<true, false, true><<<dim3(DFFp / 64, S / 64, 1), 128, 0, stream>>>(
          mg_bf + (size_t)e * S * Dp, Dp, 0,
          w1_t + (size_t)e * DFFp * Dp, Dp, 0,
          hid_bf + (size_t)e * S * DFFp, DFFp, 0,
          b1_pad + (size_t)e * DFFp, Dp, S, cnt + e, DFFp);
      // y = hid @ W2_e + b2_e
      gemm_nt_wmma<false, false, false><<<dim3(Dp / 64, S / 64, 1), 128, 0, stream>>>(
          hid_bf + (size_t)e * S * DFFp, DFFp, 0,
          w2_t + (size_t)e * Dp * DFFp, DFFp, 0,
          y32 + (size_t)e * S * Dp, Dp, 0,
          b2_l + (size_t)e * D, DFFp, S, cnt + e, D);
    }

    scatter_add_rows<<<(S * D + 255) / 256, 256, 0, stream>>>(h, y32, slot);
  }
  (void)in_sizes; (void)n_in; (void)out_size; (void)ws_size;
}
